// FourierConvolutionAttention_82351702933551
// MI455X (gfx1250) — matrix operand order repaired and checked by reading, not compiled, not run on hardware
//
#include <hip/hip_runtime.h>
#include <hip/hip_bf16.h>
#include <math.h>

// Problem constants (reference: B=4, S=2048, E=1024, H=16, D=64)
#define Bc 4
#define Sc 2048
#define Ec 1024
#define Hc 16
#define Dc 64
#define Fc 1025
#define LN_EPS 1e-5f

typedef __attribute__((ext_vector_type(16))) _Float16 v16h;
typedef __attribute__((ext_vector_type(8)))  float    v8f;

// ---------------------------------------------------------------------------
// CDNA5 async global->LDS (ASYNCcnt) path, guarded so we always compile.
// Probe result (round 2 stderr): first param of the b128 builtin is
// 'int __vector(4) __device__ *' (addrspace(1), non-const).
// ---------------------------------------------------------------------------
typedef int v4i_ __attribute__((vector_size(16)));

#if __has_builtin(__builtin_amdgcn_global_load_async_to_lds_b128) && \
    __has_builtin(__builtin_amdgcn_s_wait_asynccnt)
#define HAVE_ASYNC_LDS 1
#define GPTR(p) ((__attribute__((address_space(1))) v4i_*)(p))
#define LPTR(p) ((__attribute__((address_space(3))) v4i_*)(p))
#define ASYNC_COPY_B128(g, l) \
  __builtin_amdgcn_global_load_async_to_lds_b128(GPTR(g), LPTR(l), 0, 0)
#define ASYNC_WAIT() __builtin_amdgcn_s_wait_asynccnt(0)
#else
#define HAVE_ASYNC_LDS 0
#define ASYNC_WAIT()
#endif

// ---------------------------------------------------------------------------
// Block reductions (wave32; 256 threads = 8 waves)
// ---------------------------------------------------------------------------
__device__ __forceinline__ float block_reduce(float v, float* sred, int tid, bool do_max) {
#pragma unroll
  for (int o = 16; o > 0; o >>= 1) {
    float other = __shfl_down(v, o, 32);
    v = do_max ? fmaxf(v, other) : (v + other);
  }
  int wid = tid >> 5;
  if ((tid & 31) == 0) sred[wid] = v;
  __syncthreads();
  if (tid < 8) {
    v = sred[tid];
#pragma unroll
    for (int o = 4; o > 0; o >>= 1) {
      float other = __shfl_down(v, o, 32);
      v = do_max ? fmaxf(v, other) : (v + other);
    }
    if (tid == 0) sred[0] = v;
  }
  __syncthreads();
  float r = sred[0];
  __syncthreads();
  return r;
}

// ---------------------------------------------------------------------------
// In-LDS radix-2 FFT, N = 2048, 256 threads, shared twiddle table.
// ---------------------------------------------------------------------------
#define FFT_N   2048
#define FFT_LOG 11

__device__ __forceinline__ void fft2048(float* re, float* im,
                                        const float* twr, const float* twi,
                                        bool inverse, int tid) {
  for (int i = tid; i < FFT_N; i += 256) {
    int j = (int)(__brev((unsigned)i) >> (32 - FFT_LOG));
    if (j > i) {
      float tr = re[i]; re[i] = re[j]; re[j] = tr;
      float ti = im[i]; im[i] = im[j]; im[j] = ti;
    }
  }
  __syncthreads();
  int lh = 0;
  for (int half = 1; half < FFT_N; half <<= 1, ++lh) {
    for (int t = tid; t < FFT_N / 2; t += 256) {
      int pos = t & (half - 1);
      int i   = ((t >> lh) << (lh + 1)) + pos;
      int j   = i + half;
      int widx = pos << (10 - lh);
      float wr = twr[widx];
      float wi = inverse ? -twi[widx] : twi[widx];
      float ur = re[i], ui = im[i];
      float vr = re[j], vi = im[j];
      float tr = vr * wr - vi * wi;
      float ti = vr * wi + vi * wr;
      re[i] = ur + tr; im[i] = ui + ti;
      re[j] = ur - tr; im[j] = ui - ti;
    }
    __syncthreads();
  }
}

// ---------------------------------------------------------------------------
// Kernel 1: fused QKV projection.  y[m,n] = sum_k x[m,k]*W[n,k] + b[n]
// M=8192, K=1024, N=3*1024. Block tile 128x64 (8 waves, 32x32 per wave,
// 4 WMMAs / K-step), double-buffered LDS fed by async global->LDS B128.
// f32 tiles live in LDS; f32->f16 happens at fragment gather.
// Output layout [B,H,D,S] so FFT rows are contiguous.
// ---------------------------------------------------------------------------
__global__ void __launch_bounds__(256)
gemm_qkv_wmma(const float* __restrict__ x,
              const float* __restrict__ wq, const float* __restrict__ wk,
              const float* __restrict__ wv,
              const float* __restrict__ bq, const float* __restrict__ bk,
              const float* __restrict__ bv,
              float* __restrict__ qo, float* __restrict__ ko,
              float* __restrict__ vo) {
  __shared__ float As[2][128][36];   // 36*4 = 144B row stride (16B aligned)
  __shared__ float Bs[2][64][36];

  const int tid  = threadIdx.x;
  const int lane = tid & 31;
  const int wid  = tid >> 5;
  const int mw   = wid & 3;          // 4 wave-rows of 32
  const int nw   = wid >> 2;         // 2 wave-cols of 32
  const int rloc = lane & 15;
  const int kbA  = (lane < 16) ? 0 : 8;
  const int ksB  = (lane < 16) ? 0 : 8;

  const int m0 = blockIdx.x * 128;
  const int n0 = blockIdx.y * 64;          // global over 3*E
  const int which = n0 >> 10;              // 0=q 1=k 2=v
  const int nloc0 = n0 & 1023;
  const float* W  = (which == 0) ? wq : (which == 1) ? wk : wv;
  const float* Bb = (which == 0) ? bq : (which == 1) ? bk : bv;
  float* Out      = (which == 0) ? qo : (which == 1) ? ko : vo;

  v8f acc[2][2] = {};

  auto issue = [&](int buf, int k0) {
#if HAVE_ASYNC_LDS
#pragma unroll
    for (int s2 = tid; s2 < 128 * 8; s2 += 256) {      // A: 128 rows x 128B
      int r = s2 >> 3, seg = s2 & 7;
      ASYNC_COPY_B128(x + (long)(m0 + r) * Ec + k0 + seg * 4,
                      &As[buf][r][seg * 4]);
    }
#pragma unroll
    for (int s2 = tid; s2 < 64 * 8; s2 += 256) {       // B: 64 rows x 128B
      int r = s2 >> 3, seg = s2 & 7;
      ASYNC_COPY_B128(W + (long)(nloc0 + r) * Ec + k0 + seg * 4,
                      &Bs[buf][r][seg * 4]);
    }
#else
    for (int s2 = tid; s2 < 128 * 32; s2 += 256) {
      int r = s2 >> 5, kk = s2 & 31;
      As[buf][r][kk] = x[(long)(m0 + r) * Ec + k0 + kk];
    }
    for (int s2 = tid; s2 < 64 * 32; s2 += 256) {
      int r = s2 >> 5, kk = s2 & 31;
      Bs[buf][r][kk] = W[(long)(nloc0 + r) * Ec + k0 + kk];
    }
#endif
  };

  auto compute = [&](int buf) {
    v16h af[2], bfr[2];
#pragma unroll
    for (int mi = 0; mi < 2; ++mi) {
      const float* arow = As[buf][mw * 32 + mi * 16 + rloc];
#pragma unroll
      for (int j = 0; j < 8; ++j) {
        af[mi][j]     = (_Float16)arow[kbA + j];
        af[mi][8 + j] = (_Float16)arow[kbA + 16 + j];
      }
    }
#pragma unroll
    for (int ni = 0; ni < 2; ++ni) {
      const float* brow = Bs[buf][nw * 32 + ni * 16 + rloc];
#pragma unroll
      for (int i = 0; i < 16; ++i) bfr[ni][i] = (_Float16)brow[ksB + (i < 8 ? i : i + 8)];
    }
#pragma unroll
    for (int mi = 0; mi < 2; ++mi)
#pragma unroll
      for (int ni = 0; ni < 2; ++ni)
        acc[mi][ni] = __builtin_amdgcn_wmma_f32_16x16x32_f16(
            false, af[mi], false, bfr[ni], (short)0, acc[mi][ni], false, false);
  };

  const int NK = Ec / 32;
  issue(0, 0);
  ASYNC_WAIT();
  __syncthreads();
  for (int it = 0; it < NK; ++it) {
    const int cur = it & 1;
    if (it + 1 < NK) issue(cur ^ 1, (it + 1) * 32);
    compute(cur);
    if (it + 1 < NK) { ASYNC_WAIT(); __syncthreads(); }
  }

  // C layout: lane l -> N = l%16; VGPR r -> M = r (+8 for lanes >= 16)
#pragma unroll
  for (int ni = 0; ni < 2; ++ni) {
    const int col = nloc0 + nw * 32 + ni * 16 + rloc;   // 0..1023 within E
    const float bias = Bb[col];
#pragma unroll
    for (int mi = 0; mi < 2; ++mi) {
#pragma unroll
      for (int r = 0; r < 8; ++r) {
        const int m = m0 + mw * 32 + mi * 16 + r + ((lane < 16) ? 0 : 8);
        const int b = m >> 11;
        const int s = m & (Sc - 1);
        Out[((long)(b * (Hc * Dc) + col)) * Sc + s] = acc[mi][ni][r] + bias;
      }
    }
  }
}

// ---------------------------------------------------------------------------
// Kernel 2: fused spectral attention per (b,h,d) row.
//   FFT(q + i k) -> A[f] = |Q||K|*filter^2  (phase provably cancels)
//   real-even inverse -> attn_time -> softmax -> attn
//   FFT(attn + i v) -> O[f] = ATT[f]*V[f] -> Hermitian inverse FFT -> out
// Writes f16 activation in [B,S,E] layout for the output GEMM.
// ---------------------------------------------------------------------------
__global__ void __launch_bounds__(256)
fourier_attn_kernel(const float* __restrict__ q, const float* __restrict__ k,
                    const float* __restrict__ v,
                    const float* __restrict__ freq_filter,
                    _Float16* __restrict__ outh) {
  __shared__ float sre[FFT_N];
  __shared__ float sim[FFT_N];
  __shared__ float satt[FFT_N];
  __shared__ float saux[Fc];
  __shared__ float twr[FFT_N / 2];
  __shared__ float twi[FFT_N / 2];
  __shared__ float sred[8];

  const int tid = threadIdx.x;
  const int row = blockIdx.x;           // (b*H + h)*D + d
  const int b   = row >> 10;
  const int hd  = row & 1023;           // h*64 + d
  const float ff = freq_filter[hd];
  const float filt2 = ff * ff;
  const long base = (long)row * Sc;
  const float invN = 1.0f / (float)FFT_N;

  for (int j = tid; j < FFT_N / 2; j += 256) {
    float ang = -6.283185307179586f * (float)j * invN;
    twr[j] = __cosf(ang);
    twi[j] = __sinf(ang);
  }

  // ---- FFT 1: z = q + i k ----
  for (int i = tid; i < FFT_N; i += 256) { sre[i] = q[base + i]; sim[i] = k[base + i]; }
  __syncthreads();
  fft2048(sre, sim, twr, twi, false, tid);

  for (int f = tid; f < Fc; f += 256) {
    int g = (FFT_N - f) & (FFT_N - 1);
    float zr = sre[f], zi = sim[f], gr = sre[g], gi = sim[g];
    float Qr = 0.5f * (zr + gr), Qi = 0.5f * (zi - gi);
    float Kr = 0.5f * (zi + gi), Ki = 0.5f * (gr - zr);
    saux[f] = sqrtf((Qr * Qr + Qi * Qi) * (Kr * Kr + Ki * Ki)) * filt2;
  }
  __syncthreads();

  // ---- irfft of real-even spectrum == forward DFT / N ----
  for (int i = tid; i < FFT_N; i += 256) {
    sre[i] = (i <= FFT_N / 2) ? saux[i] : saux[FFT_N - i];
    sim[i] = 0.0f;
  }
  __syncthreads();
  fft2048(sre, sim, twr, twi, false, tid);

  // ---- softmax over sequence ----
  float lmax = -3.0e38f;
  for (int i = tid; i < FFT_N; i += 256) lmax = fmaxf(lmax, sre[i] * invN);
  const float mx = block_reduce(lmax, sred, tid, true);
  float lsum = 0.0f;
  for (int i = tid; i < FFT_N; i += 256) {
    float e = __expf(sre[i] * invN - mx);
    satt[i] = e;
    lsum += e;
  }
  const float inv_sum = 1.0f / block_reduce(lsum, sred, tid, false);

  // ---- FFT 3: z = attn + i v ----
  for (int i = tid; i < FFT_N; i += 256) {
    sre[i] = satt[i] * inv_sum;
    sim[i] = v[base + i];
  }
  __syncthreads();
  fft2048(sre, sim, twr, twi, false, tid);

  for (int f = tid; f < Fc; f += 256) {
    int g = (FFT_N - f) & (FFT_N - 1);
    float zr = sre[f], zi = sim[f], gr = sre[g], gi = sim[g];
    float Ar = 0.5f * (zr + gr), Ai = 0.5f * (zi - gi);
    float Vr = 0.5f * (zi + gi), Vi = 0.5f * (gr - zr);
    satt[f] = Ar * Vr - Ai * Vi;
    saux[f] = Ar * Vi + Ai * Vr;
  }
  __syncthreads();

  for (int i = tid; i < FFT_N; i += 256) {
    if (i <= FFT_N / 2) { sre[i] = satt[i]; sim[i] = saux[i]; }
    else                { sre[i] = satt[FFT_N - i]; sim[i] = -saux[FFT_N - i]; }
  }
  __syncthreads();
  fft2048(sre, sim, twr, twi, true, tid);

  for (int s = tid; s < FFT_N; s += 256) {
    outh[((long)(b * Sc + s)) * Ec + hd] = (_Float16)(sre[s] * invN);
  }
}

// ---------------------------------------------------------------------------
// Kernel 3: output projection (A already f16).  Same tiling as kernel 1.
// ---------------------------------------------------------------------------
__global__ void __launch_bounds__(256)
gemm_o_wmma(const _Float16* __restrict__ ah, const float* __restrict__ wo,
            const float* __restrict__ bo, float* __restrict__ y) {
  __shared__ _Float16 Ah[2][128][40];  // 40*2 = 80B row stride (16B aligned)
  __shared__ float    Bs[2][64][36];

  const int tid  = threadIdx.x;
  const int lane = tid & 31;
  const int wid  = tid >> 5;
  const int mw   = wid & 3;
  const int nw   = wid >> 2;
  const int rloc = lane & 15;
  const int kbA  = (lane < 16) ? 0 : 8;
  const int ksB  = (lane < 16) ? 0 : 8;

  const int m0 = blockIdx.x * 128;
  const int n0 = blockIdx.y * 64;

  v8f acc[2][2] = {};

  auto issue = [&](int buf, int k0) {
#if HAVE_ASYNC_LDS
#pragma unroll
    for (int s2 = tid; s2 < 128 * 4; s2 += 256) {      // A: 128 rows x 64B
      int r = s2 >> 2, seg = s2 & 3;
      ASYNC_COPY_B128(ah + (long)(m0 + r) * Ec + k0 + seg * 8,
                      &Ah[buf][r][seg * 8]);
    }
#pragma unroll
    for (int s2 = tid; s2 < 64 * 8; s2 += 256) {       // B: 64 rows x 128B
      int r = s2 >> 3, seg = s2 & 7;
      ASYNC_COPY_B128(wo + (long)(n0 + r) * Ec + k0 + seg * 4,
                      &Bs[buf][r][seg * 4]);
    }
#else
    for (int s2 = tid; s2 < 128 * 32; s2 += 256) {
      int r = s2 >> 5, kk = s2 & 31;
      Ah[buf][r][kk] = ah[(long)(m0 + r) * Ec + k0 + kk];
    }
    for (int s2 = tid; s2 < 64 * 32; s2 += 256) {
      int r = s2 >> 5, kk = s2 & 31;
      Bs[buf][r][kk] = wo[(long)(n0 + r) * Ec + k0 + kk];
    }
#endif
  };

  auto compute = [&](int buf) {
    v16h af[2], bfr[2];
#pragma unroll
    for (int mi = 0; mi < 2; ++mi) {
      const _Float16* arow = Ah[buf][mw * 32 + mi * 16 + rloc];
#pragma unroll
      for (int j = 0; j < 8; ++j) {
        af[mi][j]     = arow[kbA + j];
        af[mi][8 + j] = arow[kbA + 16 + j];
      }
    }
#pragma unroll
    for (int ni = 0; ni < 2; ++ni) {
      const float* brow = Bs[buf][nw * 32 + ni * 16 + rloc];
#pragma unroll
      for (int i = 0; i < 16; ++i) bfr[ni][i] = (_Float16)brow[ksB + (i < 8 ? i : i + 8)];
    }
#pragma unroll
    for (int mi = 0; mi < 2; ++mi)
#pragma unroll
      for (int ni = 0; ni < 2; ++ni)
        acc[mi][ni] = __builtin_amdgcn_wmma_f32_16x16x32_f16(
            false, af[mi], false, bfr[ni], (short)0, acc[mi][ni], false, false);
  };

  const int NK = Ec / 32;
  issue(0, 0);
  ASYNC_WAIT();
  __syncthreads();
  for (int it = 0; it < NK; ++it) {
    const int cur = it & 1;
    if (it + 1 < NK) issue(cur ^ 1, (it + 1) * 32);
    compute(cur);
    if (it + 1 < NK) { ASYNC_WAIT(); __syncthreads(); }
  }

#pragma unroll
  for (int ni = 0; ni < 2; ++ni) {
    const int col = n0 + nw * 32 + ni * 16 + rloc;
    const float bias = bo[col];
#pragma unroll
    for (int mi = 0; mi < 2; ++mi) {
#pragma unroll
      for (int r = 0; r < 8; ++r) {
        const int m = m0 + mw * 32 + mi * 16 + r + ((lane < 16) ? 0 : 8);
        y[(long)m * Ec + col] = acc[mi][ni][r] + bias;
      }
    }
  }
}

// ---------------------------------------------------------------------------
// Kernel 4: residual + LayerNorm over E=1024 per (b,s) row.
// ---------------------------------------------------------------------------
__global__ void __launch_bounds__(256)
resid_ln_kernel(const float* __restrict__ x, const float* __restrict__ y,
                const float* __restrict__ gamma, const float* __restrict__ beta,
                float* __restrict__ out) {
  __shared__ float sred[8];
  const int m = blockIdx.x;
  const int tid = threadIdx.x;
  const float* xr = x + (long)m * Ec;
  const float* yr = y + (long)m * Ec;

  float hv[4];
  float lsum = 0.0f, lsq = 0.0f;
#pragma unroll
  for (int i = 0; i < 4; ++i) {
    int idx = tid + i * 256;
    float vv = xr[idx] + yr[idx];
    hv[i] = vv;
    lsum += vv;
    lsq  += vv * vv;
  }
  const float mean = block_reduce(lsum, sred, tid, false) * (1.0f / Ec);
  const float msq  = block_reduce(lsq,  sred, tid, false) * (1.0f / Ec);
  const float rs   = rsqrtf(msq - mean * mean + LN_EPS);
#pragma unroll
  for (int i = 0; i < 4; ++i) {
    int idx = tid + i * 256;
    out[(long)m * Ec + idx] = (hv[i] - mean) * rs * gamma[idx] + beta[idx];
  }
}

// ---------------------------------------------------------------------------
// Launcher
// ---------------------------------------------------------------------------
extern "C" void kernel_launch(void* const* d_in, const int* in_sizes, int n_in,
                              void* d_out, int out_size, void* d_ws, size_t ws_size,
                              hipStream_t stream) {
  const float* x   = (const float*)d_in[0];
  const float* wq  = (const float*)d_in[1];
  const float* bq  = (const float*)d_in[2];
  const float* wk  = (const float*)d_in[3];
  const float* bk  = (const float*)d_in[4];
  const float* wv  = (const float*)d_in[5];
  const float* bv  = (const float*)d_in[6];
  const float* wo  = (const float*)d_in[7];
  const float* bo  = (const float*)d_in[8];
  const float* ffl = (const float*)d_in[9];
  // d_in[10] = phase_shift: provably unused (|e^{i*phase}| == 1 cancels in the
  // magnitude). d_in[11]/[12] = ln_gamma/ln_beta.
  const float* lng = (const float*)d_in[11];
  const float* lnb = (const float*)d_in[12];
  float* out = (float*)d_out;

  const long ROWS = (long)Bc * Hc * Dc;        // 4096
  const long RELT = ROWS * Sc;                 // 8M elements per tensor

  float* ws   = (float*)d_ws;
  float* q    = ws;                            // 32 MB
  float* k    = ws + RELT;                     // 32 MB
  float* v    = ws + 2 * RELT;                 // 32 MB
  _Float16* outh = (_Float16*)(ws + 3 * RELT); // 16 MB
  float* y    = ws;                            // aliases q (free after kernel 2)

  // 1) QKV projection: M=8192, Ntotal=3072, 128x64 block tiles
  dim3 g1(Bc * Sc / 128, 3 * Ec / 64);
  gemm_qkv_wmma<<<g1, 256, 0, stream>>>(x, wq, wk, wv, bq, bk, bv, q, k, v);

  // 2) fused spectral attention, one block per (b,h,d) row
  fourier_attn_kernel<<<dim3((unsigned)ROWS), 256, 0, stream>>>(q, k, v, ffl, outh);

  // 3) output projection
  dim3 g3(Bc * Sc / 128, Ec / 64);
  gemm_o_wmma<<<g3, 256, 0, stream>>>(outh, wo, bo, y);

  // 4) residual + LayerNorm
  resid_ln_kernel<<<dim3(Bc * Sc), 256, 0, stream>>>(x, y, lng, lnb, out);
}
